// Nucleus1SpatialAttention_1829656068444
// MI455X (gfx1250) — matrix operand order repaired and checked by reading, not compiled, not run on hardware
//
#include <hip/hip_runtime.h>
#include <hip/hip_bf16.h>
#include <math.h>

// ---------------------------------------------------------------------------
// Types
// ---------------------------------------------------------------------------
typedef __bf16 bf16_t;
typedef __attribute__((ext_vector_type(16))) __bf16 v16bf;
typedef __attribute__((ext_vector_type(8)))  __bf16 v8bf;
typedef __attribute__((ext_vector_type(4)))  __bf16 v4bf;
typedef __attribute__((ext_vector_type(8)))  float  v8f;

__device__ __forceinline__ v16bf join8(v8bf lo, v8bf hi) {
  v16bf r;
#pragma unroll
  for (int i = 0; i < 8; ++i) { r[i] = lo[i]; r[i + 8] = hi[i]; }
  return r;
}

// Problem constants
#define MROWS   16384        // b*t*h*w
#define CDIM    1536
#define NQKV    4608
#define NHEAD   16
#define HDIM    96
#define SLEN    1024
#define NPROB   256          // (b*t)*heads

// ---------------------------------------------------------------------------
// Generic bf16 WMMA GEMM:  C[M,N] = A[M,K] * B[K,N] + bias[N]
//   A_F32:  A is fp32 in memory, converted to bf16 while staging into LDS
//   OUT_F32: store float (rounded through bf16, matching reference cast path)
// Block tile 128x128, BK=32, 256 threads = 8 waves, wave tile 32x64
// (2x4 WMMA). Register double-buffering: global loads for K-step i+1 are
// issued while WMMAs for step i consume LDS; GL2 prefetch two steps ahead.
// ---------------------------------------------------------------------------
template <bool A_F32, bool OUT_F32>
__global__ __launch_bounds__(256)
void gemm_bf16_wmma(const void* __restrict__ Ap,
                    const bf16_t* __restrict__ B,
                    const bf16_t* __restrict__ bias,
                    void* __restrict__ Cp,
                    int M, int N, int K) {
  constexpr int BM = 128, BN = 128, BK = 32, PAD = 8;
  __shared__ alignas(16) bf16_t As[BM][BK + PAD];
  __shared__ alignas(16) bf16_t BsT[BN][BK + PAD];   // transposed: [n][k]

  const int tid  = threadIdx.x;
  const int lane = tid & 31;
  const int wave = tid >> 5;
  const int wm   = (wave & 3) * 32;      // wave M offset in block tile
  const int wn   = (wave >> 2) * 64;     // wave N offset in block tile
  const int l15  = lane & 15;
  const int half = lane >> 4;
  const int row0 = blockIdx.y * BM;
  const int col0 = blockIdx.x * BN;

  // staging thread mapping
  const int am = tid >> 1;               // A: row 0..127
  const int ak = (tid & 1) * 16;         // A: k offset 0 or 16
  const int bk = tid >> 3;               // B: k row 0..31
  const int bc = (tid & 7) * 16;         // B: col offset 0..112

  v8f acc[2][4] = {};

  // double-buffer registers
  float4 raf[4];
  v8bf   rab[2];
  v8bf   rb[2];

  auto loadTiles = [&](int k0) {
    if (A_F32) {
      const float* A = (const float*)Ap;
      const float4* g = (const float4*)(A + (size_t)(row0 + am) * K + k0 + ak);
      raf[0] = g[0]; raf[1] = g[1]; raf[2] = g[2]; raf[3] = g[3];
      __builtin_prefetch((const char*)g + 2 * BK * sizeof(float), 0, 1);
    } else {
      const bf16_t* A = (const bf16_t*)Ap;
      const v8bf* g = (const v8bf*)(A + (size_t)(row0 + am) * K + k0 + ak);
      rab[0] = g[0]; rab[1] = g[1];
      __builtin_prefetch((const char*)g + 2 * BK * sizeof(bf16_t), 0, 1);
    }
    const bf16_t* gB = B + (size_t)(k0 + bk) * N + col0 + bc;
    rb[0] = *(const v8bf*)gB;
    rb[1] = *(const v8bf*)(gB + 8);
    __builtin_prefetch((const char*)gB + (size_t)2 * BK * N * sizeof(bf16_t), 0, 1);
  };

  auto storeTiles = [&]() {
    if (A_F32) {
      const float* f = (const float*)raf;
      v8bf p0, p1;
#pragma unroll
      for (int i = 0; i < 8; ++i) { p0[i] = (bf16_t)f[i]; p1[i] = (bf16_t)f[8 + i]; }
      *(v8bf*)&As[am][ak]     = p0;
      *(v8bf*)&As[am][ak + 8] = p1;
    } else {
      *(v8bf*)&As[am][ak]     = rab[0];
      *(v8bf*)&As[am][ak + 8] = rab[1];
    }
#pragma unroll
    for (int j = 0; j < 8; ++j) BsT[bc + j][bk]     = rb[0][j];
#pragma unroll
    for (int j = 0; j < 8; ++j) BsT[bc + 8 + j][bk] = rb[1][j];
  };

  loadTiles(0);
  for (int k0 = 0; k0 < K; k0 += BK) {
    storeTiles();
    __syncthreads();
    if (k0 + BK < K) loadTiles(k0 + BK);   // overlap next global load w/ WMMA

    // ---- fragments from LDS ----
    v16bf aF[2], bF[4];
#pragma unroll
    for (int t = 0; t < 2; ++t) {
      const int r = wm + t * 16 + l15;
      const int base = half * 8;           // K interleave: 0-7/16-23 vs 8-15/24-31
      aF[t] = join8(*(const v8bf*)&As[r][base], *(const v8bf*)&As[r][base + 16]);
    }
#pragma unroll
    for (int t = 0; t < 4; ++t) {
      const int c  = wn + t * 16 + l15;
      const int kh = half * 8;             // K interleave: 0-7/16-23 vs 8-15/24-31
      bF[t] = join8(*(const v8bf*)&BsT[c][kh], *(const v8bf*)&BsT[c][kh + 16]);
    }
#pragma unroll
    for (int i = 0; i < 2; ++i)
#pragma unroll
      for (int j = 0; j < 4; ++j)
        acc[i][j] = __builtin_amdgcn_wmma_f32_16x16x32_bf16(
            false, aF[i], false, bF[j], (short)0, acc[i][j], false, false);
    __syncthreads();
  }

  // ---- epilogue: bias add, bf16 round, store ----
#pragma unroll
  for (int i = 0; i < 2; ++i) {
#pragma unroll
    for (int j = 0; j < 4; ++j) {
      const int cg = col0 + wn + j * 16 + l15;
      const float bv = (float)bias[cg];
#pragma unroll
      for (int r = 0; r < 8; ++r) {
        const int rg = row0 + wm + i * 16 + r + half * 8;
        const bf16_t vb = (bf16_t)(acc[i][j][r] + bv);
        if (OUT_F32) ((float*)Cp)[(size_t)rg * N + cg] = (float)vb;
        else         ((bf16_t*)Cp)[(size_t)rg * N + cg] = vb;
      }
    }
  }
}

// ---------------------------------------------------------------------------
// LayerNorm(q,k) + axial RoPE + scatter to [prob][s][96] layout; v relayout.
// One wave per (bt, head, s) row; 3 elems per lane (head_dim 96).
// ---------------------------------------------------------------------------
__global__ __launch_bounds__(256)
void ln_rope_scatter(const bf16_t* __restrict__ qkv,
                     const bf16_t* __restrict__ qg, const bf16_t* __restrict__ qb,
                     const bf16_t* __restrict__ kg, const bf16_t* __restrict__ kb,
                     bf16_t* __restrict__ q, bf16_t* __restrict__ k,
                     bf16_t* __restrict__ v) {
  const int lane = threadIdx.x & 31;
  const int wv   = threadIdx.x >> 5;
  const long row = (long)blockIdx.x * 8 + wv;       // enumerates (bt, head, s)
  const int s    = (int)(row & 1023);
  const int head = (int)((row >> 10) & 15);
  const int bt   = (int)(row >> 14);
  const int hh = s >> 5, ww = s & 31;

  const size_t src = ((size_t)bt * SLEN + s) * NQKV + (size_t)head * (3 * HDIM);
  const size_t dst = ((size_t)(bt * NHEAD + head) * SLEN + s) * HDIM;

  float xq[3], xk[3];
#pragma unroll
  for (int t = 0; t < 3; ++t) {
    const int d = lane + t * 32;
    xq[t] = (float)qkv[src + d];
    xk[t] = (float)qkv[src + HDIM + d];
    v[dst + d] = qkv[src + 2 * HDIM + d];           // v: passthrough relayout
  }

  // --- LayerNorm over 96 (shuffle tree reductions), round to bf16 like ref
  auto lnorm = [&](float* x, const bf16_t* g, const bf16_t* b) {
    float sm = x[0] + x[1] + x[2];
#pragma unroll
    for (int o = 1; o < 32; o <<= 1) sm += __shfl_xor(sm, o, 32);
    const float mu = sm * (1.0f / 96.0f);
    const float d0 = x[0] - mu, d1 = x[1] - mu, d2 = x[2] - mu;
    float vs = d0 * d0 + d1 * d1 + d2 * d2;
#pragma unroll
    for (int o = 1; o < 32; o <<= 1) vs += __shfl_xor(vs, o, 32);
    const float inv = rsqrtf(vs * (1.0f / 96.0f) + 1e-5f);
#pragma unroll
    for (int t = 0; t < 3; ++t) {
      const int d = lane + t * 32;
      const float y = (x[t] - mu) * inv * (float)g[d] + (float)b[d];
      x[t] = (float)(bf16_t)y;                      // bf16 round as in reference
    }
  };
  lnorm(xq, qg, qb);
  lnorm(xk, kg, kb);

  // --- axial RoPE on d < 64 (slots t=0,1); pair partner via shfl_xor(1)
  const float PI = 3.14159265358979323846f;
  const float posh = -1.0f + 2.0f * hh / 31.0f;
  const float posw = -1.0f + 2.0f * ww / 31.0f;
#pragma unroll
  for (int t = 0; t < 2; ++t) {
    const int d = lane + t * 32;
    const float base = PI * (1.0f + 127.0f * (float)((d & 31) >> 1) / 15.0f);
    const float f = (d < 32 ? posh : posw) * base;
    float sn, cs;
    __sincosf(f, &sn, &cs);
    const float pq = __shfl_xor(xq[t], 1, 32);
    const float pk = __shfl_xor(xk[t], 1, 32);
    const float sgn = (d & 1) ? 1.0f : -1.0f;      // even: -t[d+1]*sin, odd: +t[d-1]*sin
    xq[t] = xq[t] * cs + sgn * pq * sn;
    xk[t] = xk[t] * cs + sgn * pk * sn;
  }
#pragma unroll
  for (int t = 0; t < 3; ++t) {
    const int d = lane + t * 32;
    q[dst + d] = (bf16_t)xq[t];
    k[dst + d] = (bf16_t)xk[t];
  }
}

// ---------------------------------------------------------------------------
// Flash attention: one (bt,head) per blockIdx.y, 128 queries per block
// (8 waves x 16 queries). 32-key steps; K/V tiles staged in LDS (V transposed),
// online softmax per wave, P transposed through per-wave LDS buffer.
// Output written as [bt][q][head*96+d] bf16 (ready for the output GEMM).
// ---------------------------------------------------------------------------
__global__ __launch_bounds__(256)
void flash_attn(const bf16_t* __restrict__ q, const bf16_t* __restrict__ k,
                const bf16_t* __restrict__ v, bf16_t* __restrict__ o) {
  constexpr int KP = 104;   // Ks row pad (96+8)
  constexpr int VP = 40;    // Vt row pad (32+8)
  constexpr int PP = 40;    // Ps row pad (32+8)
  __shared__ alignas(16) bf16_t Ks[32][KP];        // [key][d]
  __shared__ alignas(16) bf16_t Vt[96][VP];        // [d][key] (transposed)
  __shared__ alignas(16) bf16_t Ps[8][16][PP];     // per-wave P transpose buffer

  const int tid = threadIdx.x, lane = tid & 31, w = tid >> 5;
  const int l15 = lane & 15, half = lane >> 4;
  const int prob = blockIdx.y;                     // bt*16 + head
  const int bt = prob >> 4, head = prob & 15;
  const int q0 = blockIdx.x * 128 + w * 16;
  const size_t pbase = (size_t)prob * SLEN * HDIM;

  // q fragments for 3 K-chunks of 32 (head_dim 96), kept in registers
  v16bf qF[3];
#pragma unroll
  for (int c = 0; c < 3; ++c) {
    const bf16_t* qp = q + pbase + (size_t)(q0 + l15) * HDIM + c * 32 + half * 8;
    qF[c] = join8(*(const v8bf*)qp, *(const v8bf*)(qp + 16));
  }

  float mo[8], li[8];
  v8f acc[6] = {};
#pragma unroll
  for (int j = 0; j < 8; ++j) { mo[j] = -3.0e38f; li[j] = 0.0f; }
  const float scale = 0.10206207261596575f;        // 1/sqrt(96)

  for (int kb = 0; kb < SLEN; kb += 32) {
    // ---- cooperative stage of K (row-major) and V (transposed) tiles ----
    {
      const int r  = tid >> 3;                     // key row 0..31
      const int cc = (tid & 7) * 12;               // d offset 0..84
      const bf16_t* kgp = k + pbase + (size_t)(kb + r) * HDIM + cc;
      const bf16_t* vgp = v + pbase + (size_t)(kb + r) * HDIM + cc;
      const v4bf* k4 = (const v4bf*)kgp;
      const v4bf* v4 = (const v4bf*)vgp;
      v4bf ka = k4[0], kc = k4[1], ke = k4[2];
      v4bf va = v4[0], vc = v4[1], ve = v4[2];
      *(v4bf*)&Ks[r][cc]     = ka;
      *(v4bf*)&Ks[r][cc + 4] = kc;
      *(v4bf*)&Ks[r][cc + 8] = ke;
#pragma unroll
      for (int j = 0; j < 4; ++j) {
        Vt[cc + j][r]     = va[j];
        Vt[cc + 4 + j][r] = vc[j];
        Vt[cc + 8 + j][r] = ve[j];
      }
    }
    __syncthreads();

    // ---- S = Q K^T for two 16-key sub-tiles ----
    v8f sv[2];
#pragma unroll
    for (int t = 0; t < 2; ++t) {
      v8f sa = {};
#pragma unroll
      for (int c = 0; c < 3; ++c) {
        const bf16_t* bp = &Ks[t * 16 + l15][c * 32 + half * 8];
        v16bf bF = join8(*(const v8bf*)bp, *(const v8bf*)(bp + 16));
        sa = __builtin_amdgcn_wmma_f32_16x16x32_bf16(
            false, qF[c], false, bF, (short)0, sa, false, false);
      }
      sv[t] = sa;
    }

    // ---- online softmax update (row reductions across 16-lane halves) ----
    float p0[8], p1[8];
#pragma unroll
    for (int j = 0; j < 8; ++j) {
      const float s0 = sv[0][j] * scale, s1 = sv[1][j] * scale;
      float mx = fmaxf(s0, s1);
#pragma unroll
      for (int ofs = 1; ofs < 16; ofs <<= 1) mx = fmaxf(mx, __shfl_xor(mx, ofs, 32));
      const float mn = fmaxf(mo[j], mx);
      const float corr = __expf(mo[j] - mn);
      const float e0 = __expf(s0 - mn), e1 = __expf(s1 - mn);
      float rs = e0 + e1;
#pragma unroll
      for (int ofs = 1; ofs < 16; ofs <<= 1) rs += __shfl_xor(rs, ofs, 32);
      li[j] = li[j] * corr + rs;
      mo[j] = mn;
      p0[j] = e0; p1[j] = e1;
#pragma unroll
      for (int f = 0; f < 6; ++f) acc[f][j] *= corr;
    }

    // ---- transpose P through per-wave LDS -> A-fragment; acc += P V ----
#pragma unroll
    for (int j = 0; j < 8; ++j) {
      const int r = j + half * 8;
      Ps[w][r][l15]      = (bf16_t)p0[j];
      Ps[w][r][16 + l15] = (bf16_t)p1[j];
    }
    {
      const bf16_t* pp = &Ps[w][l15][half * 8];
      v16bf pF = join8(*(const v8bf*)pp, *(const v8bf*)(pp + 16));
#pragma unroll
      for (int f = 0; f < 6; ++f) {
        const bf16_t* vp = &Vt[f * 16 + l15][half * 8];
        v16bf vF = join8(*(const v8bf*)vp, *(const v8bf*)(vp + 16));
        acc[f] = __builtin_amdgcn_wmma_f32_16x16x32_bf16(
            false, pF, false, vF, (short)0, acc[f], false, false);
      }
    }
    __syncthreads();
  }

  // ---- finalize: divide by row sum, store [bt][q][head*96+d] bf16 ----
  float inv[8];
#pragma unroll
  for (int j = 0; j < 8; ++j) inv[j] = 1.0f / li[j];
#pragma unroll
  for (int f = 0; f < 6; ++f) {
#pragma unroll
    for (int j = 0; j < 8; ++j) {
      const int qrow = q0 + j + half * 8;
      const int d = f * 16 + l15;
      o[((size_t)bt * SLEN + qrow) * CDIM + head * HDIM + d] =
          (bf16_t)(acc[f][j] * inv[j]);
    }
  }
}

// ---------------------------------------------------------------------------
// Host launcher
// Workspace layout (bytes):
//   qkv : 16384*4608*2 = 150,994,944
//   q   : 256*1024*96*2 =  50,331,648
//   k   : 50,331,648
//   v   : 50,331,648
//   o   : 16384*1536*2 =  50,331,648     (total ~352 MB)
// ---------------------------------------------------------------------------
extern "C" void kernel_launch(void* const* d_in, const int* in_sizes, int n_in,
                              void* d_out, int out_size, void* d_ws, size_t ws_size,
                              hipStream_t stream) {
  const float*  x     = (const float*)d_in[0];
  const bf16_t* W_in  = (const bf16_t*)d_in[1];
  const bf16_t* b_in  = (const bf16_t*)d_in[2];
  const bf16_t* W_out = (const bf16_t*)d_in[3];
  const bf16_t* b_out = (const bf16_t*)d_in[4];
  const bf16_t* q_g   = (const bf16_t*)d_in[5];
  const bf16_t* q_b   = (const bf16_t*)d_in[6];
  const bf16_t* k_g   = (const bf16_t*)d_in[7];
  const bf16_t* k_b   = (const bf16_t*)d_in[8];
  float* out = (float*)d_out;

  char* ws = (char*)d_ws;
  const size_t QKV_B = (size_t)MROWS * NQKV * sizeof(bf16_t);
  const size_t HBUF  = (size_t)NPROB * SLEN * HDIM * sizeof(bf16_t);
  bf16_t* qkv  = (bf16_t*)(ws);
  bf16_t* qbuf = (bf16_t*)(ws + QKV_B);
  bf16_t* kbuf = (bf16_t*)(ws + QKV_B + HBUF);
  bf16_t* vbuf = (bf16_t*)(ws + QKV_B + 2 * HBUF);
  bf16_t* obuf = (bf16_t*)(ws + QKV_B + 3 * HBUF);

  // 1) qkv = x(bf16) @ W_in + b_in     [16384 x 4608]
  gemm_bf16_wmma<true, false>
      <<<dim3(NQKV / 128, MROWS / 128), 256, 0, stream>>>(
          x, W_in, b_in, qkv, MROWS, NQKV, CDIM);

  // 2) LayerNorm(q,k) + RoPE + scatter to [prob][s][96]
  ln_rope_scatter<<<(NPROB * SLEN) / 8, 256, 0, stream>>>(
      qkv, q_g, q_b, k_g, k_b, qbuf, kbuf, vbuf);

  // 3) attention per (bt, head): 256 problems x (S=1024, D=96)
  flash_attn<<<dim3(SLEN / 128, NPROB), 256, 0, stream>>>(qbuf, kbuf, vbuf, obuf);

  // 4) out = attn_out @ W_out + b_out -> f32   [16384 x 1536]
  gemm_bf16_wmma<false, true>
      <<<dim3(CDIM / 128, MROWS / 128), 256, 0, stream>>>(
          obuf, W_out, b_out, out, MROWS, CDIM, CDIM);
}
